// MLP_57449482551364
// MI455X (gfx1250) — matrix operand order repaired and checked by reading, not compiled, not run on hardware
//
#include <hip/hip_runtime.h>
#include <hip/hip_bf16.h>

// ---------------------------------------------------------------------------
// MLP: out = SiLU(x @ (W1 - zp[:,None]).T + b1) @ W2.T
//   x [8192,4096] f32, W1 [8192,4096] f32, b1/zp [8192] f32, W2 [4096,8192] f32
// MI455X: memory-bound (512MB floor @ 23.3TB/s). bf16 WMMA 16x16x32, h in bf16
// workspace, L2-aware 16x16 tile-group swizzle, double-buffered LDS,
// hardware v_cvt_pk_bf16_f32 packing, async global->LDS for the bf16 h tiles.
// ---------------------------------------------------------------------------

#define DIMS 4096
#define HID  8192
#define TOK  8192

#define BM 128
#define BN 128
#define BK 32
#define LDT 40  // LDS row stride in bf16 elements (80 B => 16B-aligned frags)

typedef __attribute__((ext_vector_type(8)))  float        v8f;
typedef __attribute__((ext_vector_type(16))) __bf16       v16bf;
typedef __attribute__((ext_vector_type(4)))  unsigned int uint4v;
typedef __attribute__((ext_vector_type(2)))  unsigned int uint2v;
typedef __attribute__((ext_vector_type(4)))  float        float4v;

union FragU { uint4v u[2]; v16bf v; };

__device__ __forceinline__ unsigned short f32_to_bf16_rne(float f) {
    unsigned int u = __float_as_uint(f);
    u += 0x7FFFu + ((u >> 16) & 1u);
    return (unsigned short)(u >> 16);
}

// Hardware packed conversion: 2x f32 -> packed bf16x2 in one VALU op.
__device__ __forceinline__ unsigned int pack2_bf16(float a, float b) {
    unsigned int r;
    asm("v_cvt_pk_bf16_f32 %0, %1, %2" : "=v"(r) : "v"(a), "v"(b));
    return r;
}

__device__ __forceinline__ void store_bf16x4(unsigned short* dst, float4v v) {
    uint2v p;
    p.x = pack2_bf16(v.x, v.y);
    p.y = pack2_bf16(v.z, v.w);
    *(uint2v*)dst = p;
}

// A-fragment (16x32 bf16): lane<16 -> K {0..7,16..23}; lane>=16 -> K {8..15,24..31}
__device__ __forceinline__ v16bf load_frag_a(const unsigned short* lds, int row, int lane) {
    const int c0 = (lane < 16) ? 0 : 8;
    const unsigned short* p = lds + row * LDT + c0;
    FragU f;
    f.u[0] = *(const uint4v*)(p);
    f.u[1] = *(const uint4v*)(p + 16);
    return f.v;
}

// B-fragment (32x16 bf16), column-per-lane: lane<16 -> K {0..7,16..23}; lane>=16 -> K {8..15,24..31}
__device__ __forceinline__ v16bf load_frag_b(const unsigned short* lds, int row, int lane) {
    const int c0 = (lane < 16) ? 0 : 8;
    const unsigned short* p = lds + row * LDT + c0;
    FragU f;
    f.u[0] = *(const uint4v*)(p);
    f.u[1] = *(const uint4v*)(p + 16);
    return f.v;
}

// 16x16 tile-group swizzle: L2 working set per group = 16 row-panels +
// 16 col-panels (~64MB < 192MB L2); each input panel re-read only tiles/16 times.
__device__ __forceinline__ void tile_coords(int bid, int tiles_m, int tiles_n,
                                            int& tm, int& tn) {
    const int G = 16;
    const int per_group = G * G;
    const int gx = tiles_m / G;
    const int group = bid / per_group;
    const int in = bid - group * per_group;
    tm = (group % gx) * G + (in % G);
    tn = (group / gx) * G + (in / G);
}

__device__ __forceinline__ float silu(float v) {
    return v * __builtin_amdgcn_rcpf(1.0f + __expf(-v));   // fast v_rcp_f32
}

// ---------------------------------------------------------------------------
// GEMM1 + zp-fold + bias + SiLU: h[t,j] = SiLU(sum_k x[t,k]*(W1[j,k]-zp[j]) + b1[j])
// ---------------------------------------------------------------------------
__global__ __launch_bounds__(256) void mlp_gemm1_silu(
    const float* __restrict__ x, const float* __restrict__ W1,
    const float* __restrict__ b1, const float* __restrict__ zp,
    unsigned short* __restrict__ h) {
    __shared__ unsigned short As[2][BM * LDT];
    __shared__ unsigned short Bs[2][BN * LDT];

    const int tid  = threadIdx.x;
    const int lane = tid & 31;
    const int wave = tid >> 5;
    const int wm   = wave & 1;          // 2 waves over M (64 rows)
    const int wn   = wave >> 1;         // 4 waves over N (32 cols)

    int tm, tn;
    tile_coords((int)blockIdx.x, TOK / BM, HID / BN, tm, tn);
    const int mBase = tm * BM;
    const int nBase = tn * BN;

    const int lr = tid >> 3;            // 0..31
    const int lc = (tid & 7) * 4;       // 0,4,...,28

    float zpv[4];
#pragma unroll
    for (int p = 0; p < 4; ++p) zpv[p] = zp[nBase + lr + p * 32];

    const float* xg  = x  + (size_t)(mBase + lr) * DIMS + lc;
    const float* w1g = W1 + (size_t)(nBase + lr) * DIMS + lc;

    v8f acc[4][2] = {};

    // ---- prologue: stage tile k=0 into buffer 0
    float4v a[4], b[4];
#pragma unroll
    for (int p = 0; p < 4; ++p) {
        a[p] = *(const float4v*)(xg  + (size_t)p * 32 * DIMS);
        b[p] = *(const float4v*)(w1g + (size_t)p * 32 * DIMS);
    }
#pragma unroll
    for (int p = 0; p < 4; ++p) {
        store_bf16x4(&As[0][(lr + p * 32) * LDT + lc], a[p]);
        float4v t;
        t.x = b[p].x - zpv[p]; t.y = b[p].y - zpv[p];
        t.z = b[p].z - zpv[p]; t.w = b[p].w - zpv[p];
        store_bf16x4(&Bs[0][(lr + p * 32) * LDT + lc], t);
    }
    __syncthreads();

    int cur = 0;
    for (int k0 = 0; k0 < DIMS; k0 += BK) {
        const bool haveNext = (k0 + BK < DIMS);
        if (haveNext) {
#pragma unroll
            for (int p = 0; p < 4; ++p) {
                a[p] = *(const float4v*)(xg  + (size_t)p * 32 * DIMS + k0 + BK);
                b[p] = *(const float4v*)(w1g + (size_t)p * 32 * DIMS + k0 + BK);
            }
        }

        v16bf af[4], bf[2];
#pragma unroll
        for (int i = 0; i < 4; ++i)
            af[i] = load_frag_a(As[cur], wm * 64 + i * 16 + (lane & 15), lane);
#pragma unroll
        for (int j = 0; j < 2; ++j)
            bf[j] = load_frag_b(Bs[cur], wn * 32 + j * 16 + (lane & 15), lane);
#pragma unroll
        for (int i = 0; i < 4; ++i)
#pragma unroll
            for (int j = 0; j < 2; ++j)
                acc[i][j] = __builtin_amdgcn_wmma_f32_16x16x32_bf16(
                    false, af[i], false, bf[j], (short)0, acc[i][j], false, false);

        if (haveNext) {
#pragma unroll
            for (int p = 0; p < 4; ++p) {
                store_bf16x4(&As[cur ^ 1][(lr + p * 32) * LDT + lc], a[p]);
                float4v t;
                t.x = b[p].x - zpv[p]; t.y = b[p].y - zpv[p];
                t.z = b[p].z - zpv[p]; t.w = b[p].w - zpv[p];
                store_bf16x4(&Bs[cur ^ 1][(lr + p * 32) * LDT + lc], t);
            }
            __syncthreads();
            cur ^= 1;
        }
    }

    // ---- epilogue: +bias, SiLU, bf16 store of h
    const int nCol0 = nBase + wn * 32 + (lane & 15);
    const float bias0 = b1[nCol0];
    const float bias1 = b1[nCol0 + 16];
    const int mHalf = (lane >> 4) << 3;
#pragma unroll
    for (int i = 0; i < 4; ++i) {
        const int rowb = mBase + wm * 64 + i * 16 + mHalf;
#pragma unroll
        for (int j = 0; j < 2; ++j) {
            const int col = nCol0 + j * 16;
            const float biasv = j ? bias1 : bias0;
#pragma unroll
            for (int r = 0; r < 8; ++r) {
                float v = silu(acc[i][j][r] + biasv);
                h[(size_t)(rowb + r) * HID + col] = f32_to_bf16_rne(v);
            }
        }
    }
}

// ---------------------------------------------------------------------------
// GEMM2: out[t,d] = sum_k h[t,k] * W2[d,k]   (h bf16 via async->LDS, W2 f32->bf16)
// ---------------------------------------------------------------------------
__global__ __launch_bounds__(256) void mlp_gemm2(
    const unsigned short* __restrict__ h, const float* __restrict__ W2,
    float* __restrict__ out) {
    __shared__ unsigned short As[2][BM * LDT];
    __shared__ unsigned short Bs[2][BN * LDT];

    const int tid  = threadIdx.x;
    const int lane = tid & 31;
    const int wave = tid >> 5;
    const int wm   = wave & 1;
    const int wn   = wave >> 1;

    int tm, tn;
    tile_coords((int)blockIdx.x, TOK / BM, DIMS / BN, tm, tn);
    const int mBase = tm * BM;
    const int nBase = tn * BN;

    const int ar = tid >> 2;            // 0..63  (bf16 A: one 16B async copy/slot)
    const int ac = (tid & 3) * 8;       // 0,8,16,24
    const int lr = tid >> 3;            // 0..31  (f32 B: 4 floats per slot)
    const int lc = (tid & 7) * 4;

    const unsigned short* hg = h  + (size_t)(mBase + ar) * HID + ac;
    const float*          wg = W2 + (size_t)(nBase + lr) * HID + lc;

    // per-thread LDS byte offsets for the two async A slots in each buffer
    // (generic->LDS address: addr[31:0] is the LDS byte offset)
    unsigned int ldsA[2][2];
#pragma unroll
    for (int bufi = 0; bufi < 2; ++bufi)
#pragma unroll
        for (int p = 0; p < 2; ++p)
            ldsA[bufi][p] = (unsigned int)(uintptr_t)&As[bufi][(ar + p * 64) * LDT + ac];

    v8f acc[4][2] = {};

    // ---- prologue: async A tile k=0, staged B tile k=0 -> buffer 0
#pragma unroll
    for (int p = 0; p < 2; ++p) {
        unsigned long long ga = (unsigned long long)(uintptr_t)(hg + (size_t)p * 64 * HID);
        asm volatile("global_load_async_to_lds_b128 %0, %1, off"
                     :: "v"(ldsA[0][p]), "v"(ga) : "memory");
    }
    float4v b[4];
#pragma unroll
    for (int p = 0; p < 4; ++p) b[p] = *(const float4v*)(wg + (size_t)p * 32 * HID);
#pragma unroll
    for (int p = 0; p < 4; ++p) store_bf16x4(&Bs[0][(lr + p * 32) * LDT + lc], b[p]);
    asm volatile("s_wait_asynccnt 0" ::: "memory");
    __syncthreads();

    int cur = 0;
    for (int k0 = 0; k0 < HID; k0 += BK) {
        const bool haveNext = (k0 + BK < HID);
        if (haveNext) {
            // async copy next A tile straight into the other LDS buffer
#pragma unroll
            for (int p = 0; p < 2; ++p) {
                unsigned long long ga = (unsigned long long)(uintptr_t)
                    (hg + (size_t)p * 64 * HID + k0 + BK);
                asm volatile("global_load_async_to_lds_b128 %0, %1, off"
                             :: "v"(ldsA[cur ^ 1][p]), "v"(ga) : "memory");
            }
#pragma unroll
            for (int p = 0; p < 4; ++p)
                b[p] = *(const float4v*)(wg + (size_t)p * 32 * HID + k0 + BK);
        }

        v16bf af[4], bf[2];
#pragma unroll
        for (int i = 0; i < 4; ++i)
            af[i] = load_frag_a(As[cur], wm * 64 + i * 16 + (lane & 15), lane);
#pragma unroll
        for (int j = 0; j < 2; ++j)
            bf[j] = load_frag_b(Bs[cur], wn * 32 + j * 16 + (lane & 15), lane);
#pragma unroll
        for (int i = 0; i < 4; ++i)
#pragma unroll
            for (int j = 0; j < 2; ++j)
                acc[i][j] = __builtin_amdgcn_wmma_f32_16x16x32_bf16(
                    false, af[i], false, bf[j], (short)0, acc[i][j], false, false);

        if (haveNext) {
#pragma unroll
            for (int p = 0; p < 4; ++p)
                store_bf16x4(&Bs[cur ^ 1][(lr + p * 32) * LDT + lc], b[p]);
            asm volatile("s_wait_asynccnt 0" ::: "memory");
            __syncthreads();
            cur ^= 1;
        }
    }

    const int nCol0 = nBase + wn * 32 + (lane & 15);
    const int mHalf = (lane >> 4) << 3;
#pragma unroll
    for (int i = 0; i < 4; ++i) {
        const int rowb = mBase + wm * 64 + i * 16 + mHalf;
#pragma unroll
        for (int j = 0; j < 2; ++j) {
            const int col = nCol0 + j * 16;
#pragma unroll
            for (int r = 0; r < 8; ++r)
                out[(size_t)(rowb + r) * DIMS + col] = acc[i][j][r];
        }
    }
}

extern "C" void kernel_launch(void* const* d_in, const int* in_sizes, int n_in,
                              void* d_out, int out_size, void* d_ws, size_t ws_size,
                              hipStream_t stream) {
    (void)in_sizes; (void)n_in; (void)out_size; (void)ws_size;
    const float* x  = (const float*)d_in[0];
    const float* W1 = (const float*)d_in[1];
    const float* b1 = (const float*)d_in[2];
    const float* zp = (const float*)d_in[3];
    const float* W2 = (const float*)d_in[4];
    float* out = (float*)d_out;
    unsigned short* h = (unsigned short*)d_ws;   // TOK*HID bf16 = 128 MB

    dim3 blk(256, 1, 1);
    dim3 g1((TOK / BM) * (HID / BN), 1, 1);   // 4096 blocks, swizzled
    dim3 g2((TOK / BM) * (DIMS / BN), 1, 1);  // 2048 blocks, swizzled
    mlp_gemm1_silu<<<g1, blk, 0, stream>>>(x, W1, b1, zp, h);
    mlp_gemm2<<<g2, blk, 0, stream>>>(h, W2, out);
}
